// LSTM_76553497084243
// MI455X (gfx1250) — matrix operand order repaired and checked by reading, not compiled, not run on hardware
//
#include <hip/hip_runtime.h>
#include <hip/hip_bf16.h>

typedef __attribute__((ext_vector_type(16))) __bf16 v16bf;
typedef __attribute__((ext_vector_type(8)))  float  v8f;

constexpr int Bq = 32, Sq = 512, Dq = 1024, Hq = 1024;
constexpr int SBq = Sq * Bq;                 // 16384
constexpr size_t HSB = (size_t)Hq * SBq;     // 16,777,216

union FragAB { uint4 u[2]; v16bf v; };
struct bf4 { __bf16 a, b, c, d; };

// ---- K0a: X fp32 -> bf16 (same [B][S][D] layout) -------------------------
__global__ __launch_bounds__(256) void k_convX(const float* __restrict__ x,
                                               __bf16* __restrict__ out, int n4) {
  int i = blockIdx.x * 256 + threadIdx.x;
  if (i < n4) {
    float4 f = ((const float4*)x)[i];
    bf4 o; o.a = (__bf16)f.x; o.b = (__bf16)f.y; o.c = (__bf16)f.z; o.d = (__bf16)f.w;
    ((bf4*)out)[i] = o;
  }
}

// ---- K0b: U_g [D][H] fp32 -> Ut[g][H][D] bf16 (transpose for B-fragments)
__global__ __launch_bounds__(256) void k_convU(const float* __restrict__ u0,
                                               const float* __restrict__ u1,
                                               const float* __restrict__ u2,
                                               const float* __restrict__ u3,
                                               __bf16* __restrict__ ut) {
  __shared__ float tile[32][33];
  const float* U = (blockIdx.z == 0) ? u0 : (blockIdx.z == 1) ? u1
                 : (blockIdx.z == 2) ? u2 : u3;
  const int h0 = blockIdx.x * 32, k0 = blockIdx.y * 32;
  const int tx = threadIdx.x & 31, ty = threadIdx.x >> 5;
#pragma unroll
  for (int j = 0; j < 4; ++j)
    tile[ty + 8 * j][tx] = U[(size_t)(k0 + ty + 8 * j) * Hq + h0 + tx];
  __syncthreads();
  __bf16* dst = ut + (size_t)blockIdx.z * Hq * Dq;
#pragma unroll
  for (int j = 0; j < 4; ++j) {
    int hh = ty + 8 * j;
    dst[(size_t)(h0 + hh) * Dq + k0 + tx] = (__bf16)tile[tx][hh];
  }
}

// ---- K1: Gb[b][g][h] = h0[b,:] @ V_g[:,h] + bias_g[h]  (tiny GEMM) ------
__global__ __launch_bounds__(256) void k_bias(const float* __restrict__ h0p,
    const float* __restrict__ v0, const float* __restrict__ v1,
    const float* __restrict__ v2, const float* __restrict__ v3,
    const float* __restrict__ b0, const float* __restrict__ b1,
    const float* __restrict__ b2, const float* __restrict__ b3,
    float* __restrict__ Gb) {
  __shared__ float hs[8][Dq];                      // 32 KB
  const int g = blockIdx.x >> 2;
  const int h = (blockIdx.x & 3) * 256 + threadIdx.x;
  const int bbase = blockIdx.y * 8;
  for (int i = threadIdx.x; i < 8 * Dq; i += 256)
    hs[i >> 10][i & (Dq - 1)] = h0p[(size_t)(bbase + (i >> 10)) * Hq + (i & (Dq - 1))];
  __syncthreads();
  const float* Vg = (g == 0) ? v0 : (g == 1) ? v1 : (g == 2) ? v2 : v3;
  const float* Bg = (g == 0) ? b0 : (g == 1) ? b1 : (g == 2) ? b2 : b3;
  float acc[8] = {0.f, 0.f, 0.f, 0.f, 0.f, 0.f, 0.f, 0.f};
  for (int k = 0; k < Dq; ++k) {
    float v = Vg[(size_t)k * Hq + h];
#pragma unroll
    for (int j = 0; j < 8; ++j) acc[j] += hs[j][k] * v;
  }
  float bv = Bg[h];
#pragma unroll
  for (int j = 0; j < 8; ++j)
    Gb[((size_t)(bbase + j) * 4 + g) * Hq + h] = acc[j] + bv;
}

// ---- K2: fused 4-gate WMMA GEMM + forget-gate scan ----------------------
// grid = (H/64, B); 128 threads = 4 waves; wave g: 0=N(tanh),1=I,2=F,3=O.
__global__ __launch_bounds__(128) void k_lstm_wmma(
    const __bf16* __restrict__ Xbf, const __bf16* __restrict__ Ut,
    const float* __restrict__ Gb, const float* __restrict__ c0,
    float* __restrict__ Mid, float* __restrict__ outp) {
  __shared__ float gbuf[4][16][65];                // padded: conflict-free
  const int h0 = blockIdx.x * 64;
  const int b  = blockIdx.y;
  const int wave = threadIdx.x >> 5;
  const int lane = threadIdx.x & 31;
  const int lr = lane & 15;
  const int ka = (lane >> 4) * 8;    // A frag: K-runs 0-7/16-23 vs 8-15/24-31
  const int kb = (lane >> 4) * 8;    // B frag: K-runs 0-7/16-23 vs 8-15/24-31

  const __bf16* Xb = Xbf + (size_t)b * Sq * Dq;
  const __bf16* Ug = Ut + (size_t)wave * Hq * Dq;

  float gb[4];
#pragma unroll
  for (int n = 0; n < 4; ++n)
    gb[n] = Gb[((size_t)b * 4 + wave) * Hq + h0 + 16 * n + lr];

  float c = 0.f;
  if (threadIdx.x < 64) c = c0[(size_t)b * Hq + h0 + threadIdx.x];

  for (int sc = 0; sc < Sq; sc += 64) {
    v8f acc[4][4];
#pragma unroll
    for (int m = 0; m < 4; ++m)
#pragma unroll
      for (int n = 0; n < 4; ++n)
        acc[m][n] = {0.f, 0.f, 0.f, 0.f, 0.f, 0.f, 0.f, 0.f};

    for (int k = 0; k < Dq; k += 32) {
      FragAB bfr[4];
#pragma unroll
      for (int n = 0; n < 4; ++n) {
        const __bf16* p = Ug + (size_t)(h0 + 16 * n + lr) * Dq + k + kb;
        bfr[n].u[0] = *(const uint4*)p;
        bfr[n].u[1] = *(const uint4*)(p + 16);
      }
#pragma unroll
      for (int m = 0; m < 4; ++m) {
        FragAB af;
        const __bf16* p = Xb + (size_t)(sc + 16 * m + lr) * Dq + k + ka;
        af.u[0] = *(const uint4*)p;
        af.u[1] = *(const uint4*)(p + 16);
#pragma unroll
        for (int n = 0; n < 4; ++n)
          acc[m][n] = __builtin_amdgcn_wmma_f32_16x16x32_bf16(
              false, af.v, false, bfr[n].v, (short)0, acc[m][n], false, false);
      }
    }

    for (int m = 0; m < 4; ++m) {
      // each wave deposits its activated gate tile (C-layout per ISA table)
#pragma unroll
      for (int n = 0; n < 4; ++n)
#pragma unroll
        for (int r = 0; r < 8; ++r) {
          float x = acc[m][n][r] + gb[n];
          float a = (wave == 0) ? tanhf(x) : 1.f / (1.f + __expf(-x));
          gbuf[wave][r + ((lane >> 4) << 3)][(n << 4) + lr] = a;
        }
      __syncthreads();
      if (threadIdx.x < 64) {
        const int h = threadIdx.x;
#pragma unroll 4
        for (int r = 0; r < 16; ++r) {
          float Nv = gbuf[0][r][h], Iv = gbuf[1][r][h];
          float Fv = gbuf[2][r][h], Ov = gbuf[3][r][h];
          c *= Fv;                               // c_t = c0 * prod F
          float rr = c + Nv * Iv;
          float ov = tanhf(rr) * Ov;
          int s = sc + (m << 4) + r;
          Mid[((size_t)b * Sq + s) * Hq + h0 + h] = ov;
          if (s == Sq - 1) {
            outp[HSB + (size_t)b * Hq + h0 + h] = ov;                      // out_last
            outp[HSB + (size_t)Bq * Hq + (size_t)b * Hq + h0 + h] = rr;    // r_last
          }
        }
      }
      __syncthreads();
    }
  }
}

// ---- K3: Mid[b][s][h] -> output_seq[h][s*B+b] via LDS transpose ---------
__global__ __launch_bounds__(256) void k_trans(const float* __restrict__ Mid,
                                               float* __restrict__ outp) {
  __shared__ float tile[32][33];
  const int s = blockIdx.x;
  const int h0 = blockIdx.y * 32;
  const int tx = threadIdx.x & 31, ty = threadIdx.x >> 5;
#pragma unroll
  for (int j = 0; j < 4; ++j) {
    int bb = ty + 8 * j;
    tile[bb][tx] = Mid[((size_t)bb * Sq + s) * Hq + h0 + tx];
  }
  __syncthreads();
#pragma unroll
  for (int j = 0; j < 4; ++j) {
    int r = ty + 8 * j;
    outp[(size_t)(h0 + r) * SBq + (size_t)s * Bq + tx] = tile[tx][r];
  }
}

extern "C" void kernel_launch(void* const* d_in, const int* in_sizes, int n_in,
                              void* d_out, int out_size, void* d_ws, size_t ws_size,
                              hipStream_t stream) {
  (void)in_sizes; (void)n_in; (void)out_size; (void)ws_size;
  const float* inp = (const float*)d_in[0];
  const float* h0  = (const float*)d_in[1];
  const float* c0  = (const float*)d_in[2];
  // gate order in kernels: 0=l (candidate N), 1=i, 2=f, 3=u (output gate)
  const float* U[4]  = {(const float*)d_in[9],  (const float*)d_in[6],
                        (const float*)d_in[3],  (const float*)d_in[12]};
  const float* V[4]  = {(const float*)d_in[10], (const float*)d_in[7],
                        (const float*)d_in[4],  (const float*)d_in[13]};
  const float* Bv[4] = {(const float*)d_in[11], (const float*)d_in[8],
                        (const float*)d_in[5],  (const float*)d_in[14]};
  float* outp = (float*)d_out;

  char* ws = (char*)d_ws;
  __bf16* Xbf = (__bf16*)ws;                                         // 32 MB
  __bf16* Ut  = (__bf16*)(ws + (size_t)Bq * Sq * Dq * 2);            //  8 MB
  float*  Gb  = (float*)(ws + (size_t)Bq * Sq * Dq * 2 + (size_t)4 * Hq * Dq * 2);
  float*  Mid = (float*)((char*)Gb + (size_t)Bq * 4 * Hq * 4);       // 64 MB

  int n4 = Bq * Sq * Dq / 4;
  k_convX<<<n4 / 256, 256, 0, stream>>>(inp, Xbf, n4);
  k_convU<<<dim3(Hq / 32, Dq / 32, 4), 256, 0, stream>>>(U[0], U[1], U[2], U[3], Ut);
  k_bias<<<dim3(16, Bq / 8), 256, 0, stream>>>(h0, V[0], V[1], V[2], V[3],
                                               Bv[0], Bv[1], Bv[2], Bv[3], Gb);
  k_lstm_wmma<<<dim3(Hq / 64, Bq), 128, 0, stream>>>(Xbf, Ut, Gb, c0, Mid, outp);
  k_trans<<<dim3(Sq, Hq / 32), 256, 0, stream>>>(Mid, outp);
}
